// VoxelLayer_40716289966699
// MI455X (gfx1250) — matrix operand order repaired and checked by reading, not compiled, not run on hardware
//
#include <hip/hip_runtime.h>
#include <hip/hip_bf16.h>
#include <stdint.h>

// ---------------- problem constants (from reference) ----------------
#define GX 1024
#define GY 1024
#define GZ 40
#define MAP_CELLS ((long)GX * GY * GZ)       // 41,943,040
#define MAX_POINTS 32
#define MAX_VOXELS 60000
#define VX 0.1f
#define VY 0.1f
#define VZ 0.2f
#define RX (-51.2f)
#define RY (-51.2f)
#define RZ (-5.0f)

// d_out layout (floats): voxels | coors | npv | voxel_num
#define VOX_OFF   0
#define COORS_OFF (MAX_VOXELS * MAX_POINTS * 4)          // 7,680,000
#define NPV_OFF   (COORS_OFF + MAX_VOXELS * 3)           // 7,860,000
#define NUM_OFF   (NPV_OFF + MAX_VOXELS)                 // 7,920,000

typedef __attribute__((ext_vector_type(16))) _Float16 v16h;
typedef __attribute__((ext_vector_type(8)))  float    v8f;

__device__ __forceinline__ float bcast_lane(float v, int lane) {
  return __int_as_float(__builtin_amdgcn_readlane(__float_as_int(v), lane));
}

// ---------------- K1: init map / zero outputs / zero counters ----------------
__global__ void vox_init(int* __restrict__ map, long nMapInt,
                         float* __restrict__ out, long nOutFloat,
                         int* __restrict__ cnt, int nCnt) {
  long tid    = (long)blockIdx.x * blockDim.x + threadIdx.x;
  long stride = (long)gridDim.x * blockDim.x;

  int4* m4 = (int4*)map;
  long nMap4 = nMapInt >> 2;
  int4 im = make_int4(0x7FFFFFFF, 0x7FFFFFFF, 0x7FFFFFFF, 0x7FFFFFFF);
  for (long i = tid; i < nMap4; i += stride) m4[i] = im;

  float4* o4 = (float4*)out;
  long nOut4 = nOutFloat >> 2;
  float4 z4 = make_float4(0.f, 0.f, 0.f, 0.f);
  for (long i = tid; i < nOut4; i += stride) o4[i] = z4;
  for (long i = (nOut4 << 2) + tid; i < nOutFloat; i += stride) out[i] = 0.f;

  for (long i = tid; i < nCnt; i += stride) cnt[i] = 0;
}

// ---------------- K2: flat ids + representative election (atomicMin) ----------------
__global__ void vox_flat(const float4* __restrict__ pts,
                         int* __restrict__ flat, int* __restrict__ map,
                         int N, int paddedN) {
  int i = blockIdx.x * blockDim.x + threadIdx.x;
  if (i >= paddedN) return;
  int f = -1;
  if (i < N) {
    float4 p = pts[i];
    int cx = (int)floorf((p.x - RX) / VX);
    int cy = (int)floorf((p.y - RY) / VY);
    int cz = (int)floorf((p.z - RZ) / VZ);
    if (cx >= 0 && cx < GX && cy >= 0 && cy < GY && cz >= 0 && cz < GZ)
      f = (cx * GY + cy) * GZ + cz;
  }
  flat[i] = f;
  if (f >= 0) atomicMin(&map[f], i);
}

// ---------------- K3: per-tile (256 pts / wave) exclusive scan via WMMA ----------------
// Tile indicator matrix X (16x16, row-major idx = r*16+c).  Row-wise inclusive
// prefix = X * U (U upper-triangular ones) via v_wmma_f32_16x16x32_f16 (exact:
// 0/1 inputs, sums <= 256).  Cross-row offsets via v_readlane broadcasts.
// All LDS here is wave-private: no block barriers needed (in-order DS per wave;
// async staging fenced by s_wait_asynccnt).
__global__ __launch_bounds__(256) void vox_scan_wmma(
    const int* __restrict__ flat, const int* __restrict__ map,
    int* __restrict__ scanLocal, int* __restrict__ waveSums) {
  __shared__ __attribute__((aligned(16))) int flat_s[8][256];
  __shared__ __attribute__((aligned(16))) int x_s[8][256];

  const int lane = threadIdx.x & 31;
  const int wave = threadIdx.x >> 5;
  const int tile = blockIdx.x * 8 + wave;
  const int base = tile * 256;

  // gfx1250 async copy: stage this wave's 256 flat-ids into LDS (2 x b128/lane)
  {
    const int* s0 = flat + base + lane * 4;
    const int* s1 = flat + base + 128 + lane * 4;
    unsigned d0 = (unsigned)(uintptr_t)&flat_s[wave][lane * 4];
    unsigned d1 = (unsigned)(uintptr_t)&flat_s[wave][128 + lane * 4];
    asm volatile("global_load_async_to_lds_b128 %0, %1, off" :: "v"(d0), "v"(s0) : "memory");
    asm volatile("global_load_async_to_lds_b128 %0, %1, off" :: "v"(d1), "v"(s1) : "memory");
    asm volatile("s_wait_asynccnt 0" ::: "memory");
  }

  const int m16   = lane & 15;   // A-row / B-col / D-col owned by this lane
  const int hi    = lane >> 4;
  const int kbase = hi * 8;      // A: lanes 0-15 hold K=0..7, lanes 16-31 K=8..15

  // A = [X | 0]  (16x32 f16, K>=16 zero-padded)
  v16h a;
#pragma unroll
  for (int j = 0; j < 16; ++j) a[j] = (_Float16)0;
#pragma unroll
  for (int j = 0; j < 8; ++j) {
    int pos = m16 * 16 + kbase + j;
    int f   = flat_s[wave][pos];
    int gi  = base + pos;
    int ind = (f >= 0 && map[f] == gi) ? 1 : 0;  // this point is its voxel's rep
    x_s[wave][pos] = ind;
    a[j] = (_Float16)ind;
  }

  // B = [U ; 0]  (32x16 f16): elem j<8 <-> K=8*hi+j (same as A), elems 8-15 K>=16 zero, col=m16
  v16h b;
#pragma unroll
  for (int j = 0; j < 16; ++j)
    b[j] = (_Float16)((j < 8 && kbase + j <= m16) ? 1 : 0);

  v8f c = {0.f, 0.f, 0.f, 0.f, 0.f, 0.f, 0.f, 0.f};
  v8f d = __builtin_amdgcn_wmma_f32_16x16x32_f16(false, a, false, b,
                                                 (short)0, c, false, false);
  // D[r,c] = inclusive prefix within row r; lane holds (r = v + 8*hi, c = m16)

  // row totals live at c=15: lane 15 (rows 0-7) and lane 31 (rows 8-15);
  // broadcast via v_readlane (uniform, no LDS traffic)
  float t[16];
#pragma unroll
  for (int v = 0; v < 8; ++v) {
    t[v]     = bcast_lane(d[v], 15);
    t[v + 8] = bcast_lane(d[v], 31);
  }
  float offs[16];
  float run = 0.f;
#pragma unroll
  for (int r = 0; r < 16; ++r) { offs[r] = run; run += t[r]; }

#pragma unroll
  for (int v = 0; v < 8; ++v) {
    float offr = hi ? offs[v + 8] : offs[v];
    int idx = (v + hi * 8) * 16 + m16;
    float incl = d[v] + offr;
    int x = x_s[wave][idx];
    scanLocal[base + idx] = (int)(incl + 0.5f) - x;  // exclusive within tile
  }
  if (lane == 0) waveSums[tile] = (int)(run + 0.5f);
}

// ---------------- K4: scan tile sums (single block), emit voxel_num ----------------
__global__ __launch_bounds__(1024) void vox_scan_tiles(int* __restrict__ waveSums,
                                                       int T, float* __restrict__ num_out) {
  __shared__ int s[1024];
  int running = 0;
  for (int basei = 0; basei < T; basei += 1024) {
    int i = basei + (int)threadIdx.x;
    int v = (i < T) ? waveSums[i] : 0;
    s[threadIdx.x] = v;
    __syncthreads();
    for (int o = 1; o < 1024; o <<= 1) {
      int tv = (threadIdx.x >= (unsigned)o) ? s[threadIdx.x - o] : 0;
      __syncthreads();
      s[threadIdx.x] += tv;
      __syncthreads();
    }
    int incl = s[threadIdx.x];
    if (i < T) waveSums[i] = running + incl - v;  // global exclusive offset
    running += s[1023];
    __syncthreads();
  }
  if (threadIdx.x == 0)
    num_out[0] = (float)(running < MAX_VOXELS ? running : MAX_VOXELS);
}

// ---------------- K5: representatives claim voxel ids, write coors ----------------
__global__ void vox_assign(const int* __restrict__ flat,
                           const int* __restrict__ scanLocal,
                           const int* __restrict__ waveSums,
                           int* __restrict__ map,
                           float* __restrict__ coors_out, int N) {
  int i = blockIdx.x * blockDim.x + threadIdx.x;
  if (i >= N) return;
  int f = flat[i];
  if (f < 0) return;
  int m = map[f];
  if (m != i) return;  // not the representative (encoded values > N, never equal i)
  int vid = scanLocal[i] + waveSums[i >> 8];
  if (vid < MAX_VOXELS) {
    map[f] = 0x40000000 | vid;  // kept marker
    int z = f % GZ;
    int y = (f / GZ) % GY;
    int x = f / (GZ * GY);
    coors_out[(long)vid * 3 + 0] = (float)z;
    coors_out[(long)vid * 3 + 1] = (float)y;
    coors_out[(long)vid * 3 + 2] = (float)x;
  } else {
    map[f] = 0x20000000;        // dropped-voxel marker (bit30 clear)
  }
}

// ---------------- K6: scatter points into voxels ----------------
__global__ void vox_scatter(const float4* __restrict__ pts,
                            const int* __restrict__ flat,
                            const int* __restrict__ map,
                            int* __restrict__ cnt,
                            float* __restrict__ vox_out, int N) {
  int i = blockIdx.x * blockDim.x + threadIdx.x;
  if (i >= N) return;
  __builtin_prefetch((const void*)(pts + i + 4096), 0, 0);  // global_prefetch (speculative)
  int f = flat[i];
  if (f < 0) return;
  int m = map[f];
  if (!(m & 0x40000000)) return;
  int vid = m & 0x3FFFFFFF;
  int rank = atomicAdd(&cnt[vid], 1);
  if (rank < MAX_POINTS) {
    float4 p = pts[i];
    float4* dst = (float4*)(vox_out + ((long)vid * MAX_POINTS + rank) * 4);
    *dst = p;
  }
}

// ---------------- K7: per-voxel point counts ----------------
__global__ void vox_npv(const int* __restrict__ cnt, float* __restrict__ npv_out) {
  int i = blockIdx.x * blockDim.x + threadIdx.x;
  if (i < MAX_VOXELS) {
    int c = cnt[i];
    npv_out[i] = (float)(c < MAX_POINTS ? c : MAX_POINTS);
  }
}

// ---------------- launch ----------------
extern "C" void kernel_launch(void* const* d_in, const int* in_sizes, int n_in,
                              void* d_out, int out_size, void* d_ws, size_t ws_size,
                              hipStream_t stream) {
  const float4* pts = (const float4*)d_in[0];
  const int N = in_sizes[0] / 4;

  const int numTiles    = (N + 255) / 256;
  const int blocksK3    = (numTiles + 7) / 8;
  const int paddedTiles = blocksK3 * 8;
  const int paddedN     = paddedTiles * 256;

  // workspace carve-up (~178 MB): map | flat | scanLocal | waveSums | cnt
  char* ws = (char*)d_ws;
  size_t off = 0;
  int* map = (int*)(ws + off);
  off += (size_t)MAP_CELLS * 4;            off = (off + 255) & ~(size_t)255;
  int* flat = (int*)(ws + off);
  off += (size_t)paddedN * 4;              off = (off + 255) & ~(size_t)255;
  int* scanLocal = (int*)(ws + off);
  off += (size_t)paddedN * 4;              off = (off + 255) & ~(size_t)255;
  int* waveSums = (int*)(ws + off);
  off += (size_t)paddedTiles * 4;          off = (off + 255) & ~(size_t)255;
  int* cnt = (int*)(ws + off);
  off += (size_t)MAX_VOXELS * 4;
  (void)ws_size;

  float* out       = (float*)d_out;
  float* vox_out   = out + VOX_OFF;
  float* coors_out = out + COORS_OFF;
  float* npv_out   = out + NPV_OFF;
  float* num_out   = out + NUM_OFF;

  vox_init<<<2048, 256, 0, stream>>>(map, MAP_CELLS, out, (long)out_size,
                                     cnt, MAX_VOXELS);
  vox_flat<<<paddedN / 256, 256, 0, stream>>>(pts, flat, map, N, paddedN);
  vox_scan_wmma<<<blocksK3, 256, 0, stream>>>(flat, map, scanLocal, waveSums);
  vox_scan_tiles<<<1, 1024, 0, stream>>>(waveSums, paddedTiles, num_out);
  vox_assign<<<(N + 255) / 256, 256, 0, stream>>>(flat, scanLocal, waveSums, map,
                                                  coors_out, N);
  vox_scatter<<<(N + 255) / 256, 256, 0, stream>>>(pts, flat, map, cnt, vox_out, N);
  vox_npv<<<(MAX_VOXELS + 255) / 256, 256, 0, stream>>>(cnt, npv_out);
}
